// SinkhornOTLoss_62921270886720
// MI455X (gfx1250) — matrix operand order repaired and checked by reading, not compiled, not run on hardware
//
#include <hip/hip_runtime.h>

// ---------------------------------------------------------------------------
// Sinkhorn OT loss for MI455X (gfx1250, wave32, WMMA).
//   K   = exp(-20*M)  stored once in bf16 (fits in 192MB L2), plus bf16 K^T.
//   100 Sinkhorn iterations, each = two WMMA matvec passes (y = K^T u, z = K v)
//   with deterministic split-K partials, device-side convergence flag
//   mirroring the reference's err check at cpt2 % 50 == 1 (cpt2 = 1, 51).
//   loss = 100 * sum(u K v M), two-stage deterministic reduction.
//   Vector operand staged via async global->LDS (ASYNCcnt); A-tile loads
//   double-buffered (2-deep) around the WMMA to avoid rotation moves and
//   WMMA->VALU hazard nops.
// ---------------------------------------------------------------------------

#define TDIM   1024
#define VDIM   50257
#define VP     50432        // V padded: multiple of 128 (and 64); pad cols K=0,b=0
#define ALPHA  20.0f
#define W_ECR  100.0f
#define MAX_IT 100
#define STOPTHR 0.005f
#define EPSV   1e-16f

#define COL_SPLIT 4
#define ROW_SPLIT 64
#define COL_CPS   ((TDIM / 32 + COL_SPLIT - 1) / COL_SPLIT)   // 8 chunks
#define ROW_CPS   ((VP   / 32 + ROW_SPLIT - 1) / ROW_SPLIT)   // 25 chunks
#define MAX_CPS   ROW_CPS                                      // 25 -> 800 elems

typedef __attribute__((ext_vector_type(4)))  __bf16 v4bf;
typedef __attribute__((ext_vector_type(8)))  __bf16 v8bf;
typedef __attribute__((ext_vector_type(16))) __bf16 v16bf;
typedef __attribute__((ext_vector_type(8)))  float  v8f;

// ---------------------------------------------------------------------------
// Build K (bf16, padded row-major T x VP) from M (f32, T x V).
// ---------------------------------------------------------------------------
__global__ __launch_bounds__(256)
void build_k(const float* __restrict__ M, __bf16* __restrict__ Kb)
{
    const int perRow = VP / 8;
    const int idx = blockIdx.x * blockDim.x + threadIdx.x;
    const int t  = idx / perRow;
    const int c0 = (idx - t * perRow) * 8;
    if (t >= TDIM) return;
    const float* mrow = M + (size_t)t * VDIM;
    v8bf out;
#pragma unroll
    for (int j = 0; j < 8; ++j) {
        const int c = c0 + j;
        float k = 0.f;
        if (c < VDIM) k = __expf(-ALPHA * mrow[c]);   // e^-20..1 : bf16-safe, fp8 underflows
        out[j] = (__bf16)k;
    }
    *(v8bf*)(Kb + (size_t)t * VP + c0) = out;
}

// ---------------------------------------------------------------------------
// K (T x VP) -> K^T (VP x T), 64x64 tiles through LDS (coalesced both sides).
// ---------------------------------------------------------------------------
__global__ __launch_bounds__(256)
void transpose_k(const __bf16* __restrict__ Kb, __bf16* __restrict__ KT)
{
    __shared__ __bf16 tile[64][68];                    // 68*2 = 136B row pitch
    const int tx = threadIdx.x & 15;
    const int ty = threadIdx.x >> 4;
    const int v0 = blockIdx.x * 64;
    const int t0 = blockIdx.y * 64;
#pragma unroll
    for (int i = 0; i < 4; ++i) {
        const int r = ty + 16 * i;                     // local t
        *(v4bf*)&tile[r][tx * 4] =
            *(const v4bf*)(Kb + (size_t)(t0 + r) * VP + v0 + tx * 4);
    }
    __syncthreads();
#pragma unroll
    for (int i = 0; i < 4; ++i) {
        const int r = ty + 16 * i;                     // local v
        v4bf o;
#pragma unroll
        for (int j = 0; j < 4; ++j) o[j] = tile[tx * 4 + j][r];
        *(v4bf*)(KT + (size_t)(v0 + r) * TDIM + t0 + tx * 4) = o;
    }
}

__global__ void init_uv(float* __restrict__ u, __bf16* __restrict__ ub,
                        int* __restrict__ flag)
{
    const int i = blockIdx.x * blockDim.x + threadIdx.x;
    if (i < TDIM) { u[i] = 1.0f / TDIM; ub[i] = (__bf16)(1.0f / TDIM); }
    if (i == 0) flag[0] = 1;
}

// ---------------------------------------------------------------------------
// WMMA matvec: part[split][r] = sum_{k in split} Mat[r,k] * x[k]
// Mat row-major R x C (multiples of 32; R multiple of 128).
// Wave handles 16 rows; A tile = Mat rows (contiguous 16B loads per the
// 16-bit 16x32 A layout). x-slice is staged once per block into LDS via
// async global->LDS (ASYNCcnt), then B is built from conflict-free LDS
// broadcast reads (32B per lane matching the 32x16 B layout: h=0 -> K 0..7 | 16..23,
// h=1 -> K 8..15 | 24..31). A-tile loads are 2-deep double
// buffered: each WMMA consumes one buffer while global_load_b128 fills the
// other, so no rotation moves / hazard nops are needed.
// D[m,n] is N-redundant; lanes 0 and 16 extract the 16 row partials.
// ---------------------------------------------------------------------------
__global__ __launch_bounds__(256)
void matvec16_wmma(const __bf16* __restrict__ Mat, const __bf16* __restrict__ x,
                   float* __restrict__ part, int R, int C, int cps,
                   const int* __restrict__ flag)
{
    if (flag[0] == 0) return;                          // uniform: EXEC stays all-1s

    __shared__ alignas(16) __bf16 xs[MAX_CPS * 32];    // <= 1600 B

    const int tid  = threadIdx.x;
    const int wave = tid >> 5;
    const int lane = tid & 31;
    const int m    = lane & 15;
    const int h    = lane >> 4;
    const int row0 = blockIdx.x * 128 + wave * 16;
    const int nChunks = C >> 5;
    const int kc0 = blockIdx.y * cps;
    int kc1 = kc0 + cps; if (kc1 > nChunks) kc1 = nChunks;
    const int nCh = (kc1 > kc0) ? (kc1 - kc0) : 0;

    // ---- async stage x[kc0*32 .. kc1*32) into LDS (16B per lane) ----
    const int nPieces = nCh * 4;                       // 16B pieces
    if (tid < nPieces) {
        unsigned lds_off = (unsigned)(size_t)(&xs[0]) + (unsigned)tid * 16u;
        unsigned long long gaddr =
            (unsigned long long)(const void*)(x + (size_t)kc0 * 32) + (unsigned long long)tid * 16ull;
        asm volatile("global_load_async_to_lds_b128 %0, %1, off"
                     :: "v"(lds_off), "v"(gaddr) : "memory");
    }
    asm volatile("s_wait_asynccnt 0x0" ::: "memory");
    __syncthreads();

    v8f acc = {0.f, 0.f, 0.f, 0.f, 0.f, 0.f, 0.f, 0.f};
    const __bf16* arow = Mat + (size_t)(row0 + m) * (size_t)C;

    if (nCh > 0) {
        // A per lane: h=0 -> K 0..7 | 16..23 ; h=1 -> K 8..15 | 24..31
        const int ho = 8 * h;
        const __bf16* ap0 = arow + (size_t)kc0 * 32;
        v8bf lo0 = *(const v8bf*)(ap0 + ho);
        v8bf hi0 = *(const v8bf*)(ap0 + 16 + ho);
        v8bf lo1, hi1;
        int i = 0;
        while (true) {
            // ---- even chunk: consume buf0, fill buf1 ----
            {
                int kn = kc0 + i + 1; if (kn >= kc1) kn = kc0;  // clamped preload
                const __bf16* apn = arow + (size_t)kn * 32;
                lo1 = *(const v8bf*)(apn + ho);
                hi1 = *(const v8bf*)(apn + 16 + ho);
                __builtin_prefetch(arow + (size_t)(kc0 + i) * 32 + 128, 0, 1);
                v16bf A = __builtin_shufflevector(lo0, hi0,
                            0, 1, 2, 3, 4, 5, 6, 7, 8, 9, 10, 11, 12, 13, 14, 15);
                v16bf B = __builtin_shufflevector(*(const v8bf*)(xs + i * 32 + ho), *(const v8bf*)(xs + i * 32 + 16 + ho),
                            0, 1, 2, 3, 4, 5, 6, 7, 8, 9, 10, 11, 12, 13, 14, 15);
                acc = __builtin_amdgcn_wmma_f32_16x16x32_bf16(
                          false, A, false, B, (short)0, acc, false, false);
            }
            if (++i >= nCh) break;
            // ---- odd chunk: consume buf1, fill buf0 ----
            {
                int kn = kc0 + i + 1; if (kn >= kc1) kn = kc0;
                const __bf16* apn = arow + (size_t)kn * 32;
                lo0 = *(const v8bf*)(apn + ho);
                hi0 = *(const v8bf*)(apn + 16 + ho);
                __builtin_prefetch(arow + (size_t)(kc0 + i) * 32 + 128, 0, 1);
                v16bf A = __builtin_shufflevector(lo1, hi1,
                            0, 1, 2, 3, 4, 5, 6, 7, 8, 9, 10, 11, 12, 13, 14, 15);
                v16bf B = __builtin_shufflevector(*(const v8bf*)(xs + i * 32 + ho), *(const v8bf*)(xs + i * 32 + 16 + ho),
                            0, 1, 2, 3, 4, 5, 6, 7, 8, 9, 10, 11, 12, 13, 14, 15);
                acc = __builtin_amdgcn_wmma_f32_16x16x32_bf16(
                          false, A, false, B, (short)0, acc, false, false);
            }
            if (++i >= nCh) break;
        }
    }
    if (m == 0) {                                      // lanes 0 (rows 0..7), 16 (rows 8..15)
        float* dst = part + (size_t)blockIdx.y * (size_t)R + row0 + 8 * h;
#pragma unroll
        for (int r = 0; r < 8; ++r) dst[r] = acc[r];   // acc[r] = D[M=r(+8h), N=0]
    }
}

// v = b / (K^T u + eps) ; also bf16 copy for the next WMMA pass
__global__ __launch_bounds__(256)
void make_v(const float* __restrict__ ypart, float* __restrict__ v,
            __bf16* __restrict__ vb, const int* __restrict__ flag)
{
    if (flag[0] == 0) return;
    const int i = blockIdx.x * blockDim.x + threadIdx.x;
    if (i >= VP) return;
    float y = 0.f;
#pragma unroll
    for (int s = 0; s < COL_SPLIT; ++s) y += ypart[(size_t)s * VP + i];
    const float b = (i < VDIM) ? (1.0f / VDIM) : 0.f;
    const float vi = b / (y + EPSV);                   // pad: 0/(0+eps) = 0
    v[i] = vi; vb[i] = (__bf16)vi;
}

// u = a / (K v + eps)
__global__ __launch_bounds__(256)
void make_u(const float* __restrict__ zpart, float* __restrict__ u,
            __bf16* __restrict__ ub, const int* __restrict__ flag)
{
    if (flag[0] == 0) return;
    const int i = blockIdx.x * blockDim.x + threadIdx.x;
    if (i >= TDIM) return;
    float z = 0.f;
    for (int s = 0; s < ROW_SPLIT; ++s) z += zpart[(size_t)s * TDIM + i];
    const float ui = (1.0f / TDIM) / (z + EPSV);
    u[i] = ui; ub[i] = (__bf16)ui;
}

// err = sum_v |v * (K^T u) - b| ; flag = err > stopThr   (fixed-order reduction)
__global__ __launch_bounds__(256)
void err_update(const float* __restrict__ ypart, const float* __restrict__ v,
                int* __restrict__ flag)
{
    if (flag[0] == 0) return;
    __shared__ float red[256];
    float s = 0.f;
    for (int i = threadIdx.x; i < VP; i += 256) {
        float y = 0.f;
#pragma unroll
        for (int sp = 0; sp < COL_SPLIT; ++sp) y += ypart[(size_t)sp * VP + i];
        const float b = (i < VDIM) ? (1.0f / VDIM) : 0.f;
        s += fabsf(v[i] * y - b);
    }
    red[threadIdx.x] = s;
    __syncthreads();
    for (int off = 128; off > 0; off >>= 1) {
        if (threadIdx.x < off) red[threadIdx.x] += red[threadIdx.x + off];
        __syncthreads();
    }
    if (threadIdx.x == 0) flag[0] = (red[0] > STOPTHR) ? 1 : 0;
}

// loss partials: sum u[t]*K[t,c]*v[c]*M[t,c]  (always runs, uses final u,v)
__global__ __launch_bounds__(256)
void loss_partial(const float* __restrict__ M, const __bf16* __restrict__ Kb,
                  const float* __restrict__ u, const float* __restrict__ v,
                  float* __restrict__ part)
{
    __shared__ float red[256];
    const int perRow = VP / 8;
    const size_t total = (size_t)TDIM * perRow;
    float s = 0.f;
    for (size_t idx = (size_t)blockIdx.x * blockDim.x + threadIdx.x; idx < total;
         idx += (size_t)gridDim.x * blockDim.x) {
        const int t  = (int)(idx / perRow);
        const int c0 = (int)(idx - (size_t)t * perRow) * 8;
        const float ut = u[t];
        const float*  mrow = M  + (size_t)t * VDIM;
        const __bf16* krow = Kb + (size_t)t * VP;
#pragma unroll
        for (int j = 0; j < 8; ++j) {
            const int c = c0 + j;
            if (c < VDIM) s += ut * (float)krow[c] * v[c] * mrow[c];
        }
    }
    red[threadIdx.x] = s;
    __syncthreads();
    for (int off = 128; off > 0; off >>= 1) {
        if (threadIdx.x < off) red[threadIdx.x] += red[threadIdx.x + off];
        __syncthreads();
    }
    if (threadIdx.x == 0) part[blockIdx.x] = red[0];
}

__global__ __launch_bounds__(256)
void loss_final(const float* __restrict__ part, int n, float* __restrict__ out)
{
    __shared__ float red[256];
    float s = 0.f;
    for (int i = threadIdx.x; i < n; i += 256) s += part[i];
    red[threadIdx.x] = s;
    __syncthreads();
    for (int off = 128; off > 0; off >>= 1) {
        if (threadIdx.x < off) red[threadIdx.x] += red[threadIdx.x + off];
        __syncthreads();
    }
    if (threadIdx.x == 0) out[0] = W_ECR * red[0];
}

// ---------------------------------------------------------------------------
extern "C" void kernel_launch(void* const* d_in, const int* in_sizes, int n_in,
                              void* d_out, int out_size, void* d_ws, size_t ws_size,
                              hipStream_t stream)
{
    const float* M = (const float*)d_in[0];
    float* out = (float*)d_out;

    const int NB_LOSS = 2048;

    // workspace carve (256B aligned regions), ~208 MB total
    size_t off = 0;
    auto carve = [&](size_t bytes) {
        size_t o = off;
        off = (off + bytes + 255) & ~(size_t)255;
        return o;
    };
    char* base  = (char*)d_ws;
    size_t oKb  = carve((size_t)TDIM * VP * 2);           // K bf16
    size_t oKT  = carve((size_t)VP * TDIM * 2);           // K^T bf16
    size_t oYp  = carve((size_t)COL_SPLIT * VP * 4);      // y split partials
    size_t oZp  = carve((size_t)ROW_SPLIT * TDIM * 4);    // z split partials
    size_t oU   = carve((size_t)TDIM * 4);
    size_t oUb  = carve((size_t)TDIM * 2);
    size_t oV   = carve((size_t)VP * 4);
    size_t oVb  = carve((size_t)VP * 2);
    size_t oFlg = carve(256);
    size_t oLp  = carve((size_t)NB_LOSS * 4);
    if (ws_size < off) return;                            // workspace too small

    __bf16* Kb = (__bf16*)(base + oKb);
    __bf16* KT = (__bf16*)(base + oKT);
    float*  yp = (float*)(base + oYp);
    float*  zp = (float*)(base + oZp);
    float*  u  = (float*)(base + oU);
    __bf16* ub = (__bf16*)(base + oUb);
    float*  v  = (float*)(base + oV);
    __bf16* vb = (__bf16*)(base + oVb);
    int*  flag = (int*)(base + oFlg);
    float*  lp = (float*)(base + oLp);

    // --- setup: K, K^T, u = 1/T, flag = 1 ---
    build_k<<<(TDIM * (VP / 8) + 255) / 256, 256, 0, stream>>>(M, Kb);
    transpose_k<<<dim3(VP / 64, TDIM / 64), 256, 0, stream>>>(Kb, KT);
    init_uv<<<(TDIM + 255) / 256, 256, 0, stream>>>(u, ub, flag);

    // --- Sinkhorn: fixed 100 iterations, device flag = while-loop condition ---
    for (int it = 1; it <= MAX_IT; ++it) {
        matvec16_wmma<<<dim3(VP / 128, COL_SPLIT), 256, 0, stream>>>(
            KT, ub, yp, VP, TDIM, COL_CPS, flag);                  // y = K^T u
        make_v<<<VP / 256, 256, 0, stream>>>(yp, v, vb, flag);
        matvec16_wmma<<<dim3(TDIM / 128, ROW_SPLIT), 256, 0, stream>>>(
            Kb, vb, zp, TDIM, VP, ROW_CPS, flag);                  // z = K v
        make_u<<<(TDIM + 255) / 256, 256, 0, stream>>>(zp, u, ub, flag);
        if (it == 1 || it == 51) {                                 // cpt2 % 50 == 1
            matvec16_wmma<<<dim3(VP / 128, COL_SPLIT), 256, 0, stream>>>(
                KT, ub, yp, VP, TDIM, COL_CPS, flag);              // K^T u2
            err_update<<<1, 256, 0, stream>>>(yp, v, flag);
        }
    }

    // --- loss = 100 * sum(u K v M), deterministic two-stage reduction ---
    loss_partial<<<NB_LOSS, 256, 0, stream>>>(M, Kb, u, v, lp);
    loss_final<<<1, 256, 0, stream>>>(lp, NB_LOSS, out);
}
